// IterativePreimage_8959301779820
// MI455X (gfx1250) — matrix operand order repaired and checked by reading, not compiled, not run on hardware
//
#include <hip/hip_runtime.h>

// ---------------- problem constants (reference hardcodes these) -------------
constexpr int N_    = 4096;
constexpr int M_    = 8192;
constexpr int D_    = 1024;
constexpr int TOPK  = 32;
constexpr int ITERS = 15;

constexpr int ROWS     = 32;            // rows per workgroup == two WMMA M-tiles
constexpr int BLK_COLS = 256;           // columns per workgroup (16 waves x 16)
constexpr int NB       = M_ / BLK_COLS; // 32 column blocks
constexpr int THREADS_A = 512;          // 16 waves (wave32)
constexpr int THREADS_B = 256;

// dynamic LDS carve-up for score_topk:
//   [0,           64KB) : A tile, 32 x 1024 bf16   (GEMM path only)
//   [A_end, +32KB)      : score tile, 32 x 256 f32
constexpr size_t SMEM_A      = (size_t)ROWS * D_ * 2;          // 64KB
constexpr size_t SMEM_SCORE  = (size_t)ROWS * BLK_COLS * 4;    // 32KB
constexpr size_t SMEM_GEMM   = SMEM_A + SMEM_SCORE;            // 96KB
constexpr size_t SMEM_NOGEMM = SMEM_SCORE;                     // 32KB

typedef __bf16 bf16x8  __attribute__((ext_vector_type(8)));
typedef __bf16 bf16x16 __attribute__((ext_vector_type(16)));
typedef float  f32x8   __attribute__((ext_vector_type(8)));

static __device__ __forceinline__ unsigned short f2bf(float f) {
    unsigned int u = __float_as_uint(f);
    u += 0x7FFFu + ((u >> 16) & 1u);            // round-to-nearest-even
    return (unsigned short)(u >> 16);
}

// ---------------- y (f32) -> y_bf16 ----------------------------------------
__global__ __launch_bounds__(256)
void cvt_y_kernel(const float* __restrict__ y, unsigned short* __restrict__ yb) {
    const int i = (blockIdx.x * 256 + threadIdx.x) * 4;
    if (i < M_ * D_) {
        const float4 f = *(const float4*)&y[i];
        ushort4 b;
        b.x = f2bf(f.x); b.y = f2bf(f.y); b.z = f2bf(f.z); b.w = f2bf(f.w);
        *(ushort4*)&yb[i] = b;
    }
}

// ---------------- fused score + per-block top-32 ----------------------------
// USE_GEMM=1: scores = pre_bf16 @ y_bf16^T via v_wmma_f32_16x16x32_bf16,
//             2 row-tiles per wave sharing one B fragment (2 WMMA / B load).
// USE_GEMM=0: scores = K (initial iteration)
template <bool USE_GEMM>
__global__ __launch_bounds__(THREADS_A)
void score_topk_kernel(const float*          __restrict__ Kmat,
                       const unsigned short* __restrict__ preb,   // [N][D] bf16
                       const unsigned short* __restrict__ yb,     // [M][D] bf16
                       float*                __restrict__ cand_v, // [N][NB][TOPK]
                       int*                  __restrict__ cand_i)
{
    extern __shared__ char smem[];
    unsigned short* sA = (unsigned short*)smem;                         // GEMM only
    float (*sScore)[BLK_COLS] =
        (float (*)[BLK_COLS])(smem + (USE_GEMM ? SMEM_A : 0));

    const int tid  = threadIdx.x;
    const int lane = tid & 31;
    const int wave = tid >> 5;
    const int cb   = blockIdx.x;                     // column block   [0,32)
    const int rt   = blockIdx.y;                     // row tile       [0,128)
    const int row0 = rt * ROWS;
    const int col0 = cb * BLK_COLS;

    if (USE_GEMM) {
        // stage 32x1024 bf16 A tile into LDS (16B vector copies)
        for (int i = tid * 8; i < ROWS * D_; i += THREADS_A * 8) {
            const int r = i >> 10, d = i & 1023;
            *(uint4*)&sA[i] = *(const uint4*)&preb[(size_t)(row0 + r) * D_ + d];
        }
        __syncthreads();

        // each wave owns a 32x16 output strip at columns col0 + wave*16:
        // two 16x16 WMMA tiles (rows 0-15 and 16-31) sharing one B fragment.
        const int kg = lane >> 4;                    // K half select
        const int ar = lane & 15;                    // A row (M) within tile
        const unsigned short* ybrow =
            yb + (size_t)(col0 + wave * 16 + (lane & 15)) * D_ + kg * 16;

        f32x8 acc0 = {}, acc1 = {};
        for (int d0 = 0; d0 < D_; d0 += 32) {
            // B fragment: 16 contiguous K per lane (32B load from y_bf16 row)
            bf16x16 b = *(const bf16x16*)&ybrow[d0];
            if (d0 + 128 < D_) __builtin_prefetch(&ybrow[d0 + 128], 0, 3);
            // A fragments (rows ar and ar+16):
            // lane<16 -> K {0..15}; lane>=16 -> K {16..31}
            bf16x8 a0lo = *(const bf16x8*)&sA[ar * D_ + d0 + kg * 16];
            bf16x8 a0hi = *(const bf16x8*)&sA[ar * D_ + d0 + 8 + kg * 16];
            bf16x8 a1lo = *(const bf16x8*)&sA[(ar + 16) * D_ + d0 + kg * 16];
            bf16x8 a1hi = *(const bf16x8*)&sA[(ar + 16) * D_ + d0 + 8 + kg * 16];
            bf16x16 a0 = __builtin_shufflevector(a0lo, a0hi,
                             0,1,2,3,4,5,6,7,8,9,10,11,12,13,14,15);
            bf16x16 a1 = __builtin_shufflevector(a1lo, a1hi,
                             0,1,2,3,4,5,6,7,8,9,10,11,12,13,14,15);
            acc0 = __builtin_amdgcn_wmma_f32_16x16x32_bf16(
                       false, a0, false, b, (short)0, acc0, false, false);
            acc1 = __builtin_amdgcn_wmma_f32_16x16x32_bf16(
                       false, a1, false, b, (short)0, acc1, false, false);
        }
        // C/D layout: VGPR r -> row r + (lane>>4)*8, col lane&15
        {
            const int crow = (lane >> 4) * 8;
            const int ccol = wave * 16 + (lane & 15);
#pragma unroll
            for (int r = 0; r < 8; ++r) {
                sScore[crow + r][ccol]      = acc0[r];
                sScore[16 + crow + r][ccol] = acc1[r];
            }
        }
    } else {
        for (int i = tid; i < ROWS * BLK_COLS; i += THREADS_A) {
            const int r = i >> 8, c = i & 255;
            sScore[r][c] = Kmat[(size_t)(row0 + r) * M_ + col0 + c];
        }
    }
    __syncthreads();

    // per-row top-32 within this 256-col block; wave w handles rows w and w+16
#pragma unroll
    for (int rr = 0; rr < 2; ++rr) {
        const int srow = wave + rr * 16;
        float v[8];
#pragma unroll
        for (int s = 0; s < 8; ++s) v[s] = sScore[srow][lane + 32 * s];

        float outv = 0.f; int outc = 0;
        for (int it = 0; it < TOPK; ++it) {
            float bv = v[0]; int bs = 0;
#pragma unroll
            for (int s = 1; s < 8; ++s) if (v[s] > bv) { bv = v[s]; bs = s; }
            int bc = lane + 32 * bs;
#pragma unroll
            for (int off = 16; off >= 1; off >>= 1) {  // wave32 argmax, tie->lower col
                const float ov = __shfl_xor(bv, off);
                const int   oc = __shfl_xor(bc, off);
                if (ov > bv || (ov == bv && oc < bc)) { bv = ov; bc = oc; }
            }
            if ((bc & 31) == lane) v[bc >> 5] = -__builtin_inff();
            if (it == lane) { outv = bv; outc = bc; }
        }
        const size_t base = ((size_t)(row0 + srow) * NB + cb) * TOPK;
        cand_v[base + lane] = outv;
        cand_i[base + lane] = col0 + outc;
    }
}

// ---------------- merge candidates -> global top-32, weights, gather --------
__global__ __launch_bounds__(THREADS_B)
void merge_gather_kernel(const float* __restrict__ cand_v,
                         const int*   __restrict__ cand_i,
                         const float* __restrict__ y,        // [M][D] f32
                         float*       __restrict__ pre_out,  // [N][D] (d_out)
                         int*         __restrict__ inds_out, // [N][TOPK] (d_out tail)
                         unsigned short* __restrict__ preb)  // [N][D] bf16
{
    __shared__ int   sCol[NB * TOPK];   // 1024 candidate columns
    __shared__ float sW[TOPK];
    __shared__ int   sI[TOPK];

    const int row  = blockIdx.x;
    const int tid  = threadIdx.x;
    const int lane = tid & 31;
    const int wave = tid >> 5;
    const size_t cbase = (size_t)row * (NB * TOPK);

    for (int i = tid; i < NB * TOPK; i += THREADS_B) sCol[i] = cand_i[cbase + i];
    __syncthreads();

    if (wave == 0) {
        float v[NB];                    // entry e = lane + 32*j
#pragma unroll
        for (int j = 0; j < NB; ++j) v[j] = cand_v[cbase + lane + 32 * j];

        float selv = 0.f; int sele = 0;
        for (int it = 0; it < TOPK; ++it) {
            float bv = v[0]; int bj = 0;
#pragma unroll
            for (int j = 1; j < NB; ++j) if (v[j] > bv) { bv = v[j]; bj = j; }
            int be = lane + 32 * bj;
#pragma unroll
            for (int off = 16; off >= 1; off >>= 1) {
                const float ov = __shfl_xor(bv, off);
                const int   oe = __shfl_xor(be, off);
                if (ov > bv || (ov == bv && oe < be)) { bv = ov; be = oe; }
            }
            if ((be & 31) == lane) v[be >> 5] = -__builtin_inff();
            if (it == lane) { selv = bv; sele = be; }
        }
        // L1 normalize across the 32 selected values
        float a = fabsf(selv);
#pragma unroll
        for (int off = 16; off >= 1; off >>= 1) a += __shfl_xor(a, off);
        const float w   = selv / fmaxf(a, 1e-12f);
        const int   col = sCol[sele];
        sW[lane] = w; sI[lane] = col;
        inds_out[(size_t)row * TOPK + lane] = col;
    }
    __syncthreads();

    // weighted gather of 32 rows of y (f32); thread t -> 4 consecutive d
    const int d0 = tid * 4;
    float4 acc = {0.f, 0.f, 0.f, 0.f};
#pragma unroll 4
    for (int k = 0; k < TOPK; ++k) {
        const float4 yv = *(const float4*)&y[(size_t)sI[k] * D_ + d0];
        const float  wk = sW[k];
        acc.x += wk * yv.x; acc.y += wk * yv.y;
        acc.z += wk * yv.z; acc.w += wk * yv.w;
    }
    *(float4*)&pre_out[(size_t)row * D_ + d0] = acc;
    ushort4 pb;
    pb.x = f2bf(acc.x); pb.y = f2bf(acc.y); pb.z = f2bf(acc.z); pb.w = f2bf(acc.w);
    *(ushort4*)&preb[(size_t)row * D_ + d0] = pb;
}

// ---------------- launcher ---------------------------------------------------
extern "C" void kernel_launch(void* const* d_in, const int* in_sizes, int n_in,
                              void* d_out, int out_size, void* d_ws, size_t ws_size,
                              hipStream_t stream) {
    (void)in_sizes; (void)n_in; (void)out_size; (void)ws_size;
    const float* Kmat = (const float*)d_in[0];
    const float* y    = (const float*)d_in[1];

    float* pre_out = (float*)d_out;
    int*   inds_out = (int*)((float*)d_out + (size_t)N_ * D_);

    // workspace carve-up: y_bf16 (16MB) | pre_bf16 (8MB) | cand_v (16MB) | cand_i (16MB)
    char* ws = (char*)d_ws;
    unsigned short* yb     = (unsigned short*)ws;
    unsigned short* preb   = (unsigned short*)(ws + (size_t)M_ * D_ * 2);
    float*          cand_v = (float*)(ws + (size_t)M_ * D_ * 2 + (size_t)N_ * D_ * 2);
    int*            cand_i = (int*)((char*)cand_v + (size_t)N_ * NB * TOPK * 4);

    cvt_y_kernel<<<(M_ * D_) / (256 * 4), 256, 0, stream>>>(y, yb);

    const dim3 gA(NB, N_ / ROWS);   // 32 x 128 workgroups
    // iteration 0: top-k directly on K
    score_topk_kernel<false><<<gA, THREADS_A, SMEM_NOGEMM, stream>>>(
        Kmat, preb, yb, cand_v, cand_i);
    merge_gather_kernel<<<N_, THREADS_B, 0, stream>>>(
        cand_v, cand_i, y, pre_out, inds_out, preb);
    // 15 iterations: fused bf16 WMMA GEMM + top-k, then merge/normalize/gather
    for (int it = 0; it < ITERS; ++it) {
        score_topk_kernel<true><<<gA, THREADS_A, SMEM_GEMM, stream>>>(
            Kmat, preb, yb, cand_v, cand_i);
        merge_gather_kernel<<<N_, THREADS_B, 0, stream>>>(
            cand_v, cand_i, y, pre_out, inds_out, preb);
    }
}
